// SyntheticGraphLearner_83528523973379
// MI455X (gfx1250) — matrix operand order repaired and checked by reading, not compiled, not run on hardware
//
#include <hip/hip_runtime.h>
#include <hip/hip_bf16.h>

// Async global->LDS staging (compiled OK in round 2).
#define USE_ASYNC_LDS 1

#if defined(__has_builtin)
#  if __has_builtin(__builtin_amdgcn_tensor_load_to_lds)
#    define HAVE_TDM 1
#  endif
#endif
#ifndef HAVE_TDM
#  define HAVE_TDM 0
#endif

typedef __attribute__((ext_vector_type(16))) _Float16 v16h;
typedef __attribute__((ext_vector_type(8)))  _Float16 v8h;
typedef __attribute__((ext_vector_type(8)))  float    v8f;
typedef unsigned int u32x4 __attribute__((ext_vector_type(4)));
typedef int          i32x8 __attribute__((ext_vector_type(8)));
typedef int          i32x4 __attribute__((ext_vector_type(4)));

#define EPSV 1e-5f

// ---- per-thread 8-byte global -> LDS copy (ASYNCcnt path) ----
__device__ __forceinline__ void stage8(const _Float16* src, _Float16* dst) {
#if USE_ASYNC_LDS
  unsigned ldsoff = (unsigned)(unsigned long long)(void*)dst;   // low 32b of generic = LDS offset
  asm volatile("global_load_async_to_lds_b64 %0, %1, off" :: "v"(ldsoff), "v"(src) : "memory");
#else
  *(unsigned long long*)dst = *(const unsigned long long*)src;
#endif
}
__device__ __forceinline__ void stage_wait() {
#if USE_ASYNC_LDS
  asm volatile("s_wait_asynccnt 0x0" ::: "memory");
#endif
}
// cooperative contiguous panel copy, halfs multiple of 4*blockDim
__device__ __forceinline__ void stage_panel(const _Float16* src, _Float16* dst, int halfs) {
  for (int idx = (int)threadIdx.x * 4; idx < halfs; idx += (int)blockDim.x * 4)
    stage8(src + idx, dst + idx);
  stage_wait();
}

#if HAVE_TDM
// ---- TDM: DMA a 16 x Kp f16 panel (row stride Kp) into LDS (TENSORcnt path) ----
// 6-arg builtin form: (g0 u32x4, g1 i32x8, g2 i32x4, g3 i32x4, i32x8, cpol)
__device__ __forceinline__ void tdm_load_panel(const _Float16* gsrc, void* ldsDst, int Kp) {
  unsigned long long ga = (unsigned long long)gsrc;
  unsigned ldsoff = (unsigned)(unsigned long long)ldsDst;
  u32x4 g0;
  g0[0] = 1u;                                                 // count=1 (valid user D#)
  g0[1] = ldsoff;                                             // lds_addr (bytes)
  g0[2] = (unsigned)ga;                                       // global_addr[31:0]
  g0[3] = (unsigned)((ga >> 32) & 0x01FFFFFFu) | (2u << 30);  // global_addr[56:32] | type=2
  i32x8 g1;
  g1[0] = 1 << 16;                           // data_size = 2B; no multicast/pad/iterate
  g1[1] = (Kp & 0xFFFF) << 16;               // tensor_dim0[15:0]  (bits 79:48 span w1/w2)
  g1[2] = ((unsigned)Kp >> 16) | (16 << 16); // tensor_dim0[31:16] | tensor_dim1[15:0]=16
  g1[3] = (Kp & 0xFFFF) << 16;               // tensor_dim1[31:16]=0 | tile_dim0 = Kp
  g1[4] = 16;                                // tile_dim1 = 16, tile_dim2 = 0
  g1[5] = Kp;                                // tensor_dim0_stride[31:0] (data_size units)
  g1[6] = 0;                                 // stride hi / dim1_stride lo
  g1[7] = 0;
  i32x4 z4 = {0, 0, 0, 0};
  i32x8 z8 = {0, 0, 0, 0, 0, 0, 0, 0};
  __builtin_amdgcn_tensor_load_to_lds(g0, g1, z4, z4, z8, 0);
}
#endif

// ---------------------------------------------------------------- f32 -> f16
__global__ void k_cvt_f16(const float* __restrict__ in, _Float16* __restrict__ out, int n) {
  int i = blockIdx.x * blockDim.x + threadIdx.x;
  if (i < n) out[i] = (_Float16)in[i];
}

// f32 [Co][K] -> f16 [Mpad][Kp], zero padded (safe for DMA tile loads)
__global__ void k_cvt_wpad(const float* __restrict__ in, _Float16* __restrict__ out,
                           int Co, int K, int Kp, int Mpad) {
  int i = blockIdx.x * blockDim.x + threadIdx.x;
  if (i >= Mpad * Kp) return;
  int m = i / Kp, k = i - m * Kp;
  out[i] = (m < Co && k < K) ? (_Float16)in[m * K + k] : (_Float16)0.f;
}

// -------------------------------------------------- implicit-GEMM conv, WMMA
// Weight panel [16][Kp] staged once into LDS by the Tensor Data Mover; inner
// K-loop is barrier-free: LDS b128 A-fragments + im2col B gathers + v_wmma.
// outMode 0: raw f32 NCHW (pre-BN).  outMode 1: f16 [n][Co] with bias (conv5).
__global__ void k_conv_wmma(const _Float16* __restrict__ X, const _Float16* __restrict__ Wt,
                            const float* __restrict__ bias,
                            float* __restrict__ rawOut, _Float16* __restrict__ nkOut,
                            int outMode,
                            int Ci, int Hi, int Wi, int Co, int Kk, int stride, int pad,
                            int Ho, int Wo, int Nimg, int Kp)
{
  extern __shared__ _Float16 Wpanel[];   // 16 * Kp halfs (<= 64KB, LDS is 320KB/WGP)

  const int lane  = threadIdx.x & 31;
  const int wave  = threadIdx.x >> 5;
  const int K     = Ci * Kk * Kk;
  const int K2    = Kk * Kk;
  const int Ntot  = Nimg * Ho * Wo;

  const int mtile = blockIdx.y;
  const int nbase = (blockIdx.x * (int)(blockDim.x >> 5) + wave) << 4;

  const int  ncol   = nbase + (lane & 15);
  const bool nvalid = (ncol < Ntot);
  int img = 0, oy = 0, ox = 0;
  if (nvalid) {
    img = ncol / (Ho * Wo);
    int sp = ncol - img * (Ho * Wo);
    oy = sp / Wo;
    ox = sp - oy * Wo;
  }
  const int iy0   = oy * stride - pad;
  const int ix0   = ox * stride - pad;
  const int row   = lane & 15;                  // A row owned by this lane
  const int khalf = (lane >> 4) << 4;           // B: K half (0 or 16)
  const int a_b8  = (lane >> 4) << 4;           // A: K half (0 or 16)

  // ---- stage the whole weight panel once ----
  const _Float16* wsrc = Wt + (size_t)mtile * 16 * Kp;
#if HAVE_TDM
  if (wave == 0) {
    tdm_load_panel(wsrc, Wpanel, Kp);
    __builtin_amdgcn_s_wait_tensorcnt(0);
  }
#else
  stage_panel(wsrc, Wpanel, 16 * Kp);
#endif
  __syncthreads();

  v8f c = {};
  for (int kk = 0; kk < K; kk += 32) {
    // A fragment: two contiguous 16B runs from the LDS panel
    v8h alo = *(const v8h*)&Wpanel[row * Kp + kk + a_b8];
    v8h ahi = *(const v8h*)&Wpanel[row * Kp + kk + 8 + a_b8];
    v16h a = __builtin_shufflevector(alo, ahi, 0,1,2,3,4,5,6,7,8,9,10,11,12,13,14,15);

    // B fragment (im2col gather): 16 consecutive k per lane
    v16h b;
    int k0 = kk + khalf;
    int ci = k0 / K2;
    int r  = k0 - ci * K2;
    int ky = r / Kk;
    int kx = r - ky * Kk;
#pragma unroll
    for (int e = 0; e < 16; ++e) {
      int k = k0 + e;
      _Float16 v = (_Float16)0.f;
      if (nvalid && k < K) {
        int iy = iy0 + ky, ix = ix0 + kx;
        if (iy >= 0 && iy < Hi && ix >= 0 && ix < Wi)
          v = X[((img * Ci + ci) * Hi + iy) * Wi + ix];
      }
      b[e] = v;
      if (++kx == Kk) { kx = 0; if (++ky == Kk) { ky = 0; ++ci; } }
    }
    c = __builtin_amdgcn_wmma_f32_16x16x32_f16(false, a, false, b, (short)0, c, false, false);
  }

#pragma unroll
  for (int i = 0; i < 8; ++i) {
    int m = mtile * 16 + i + ((lane >> 4) << 3);
    if (m < Co && nvalid) {
      float y = c[i] + bias[m];
      if (outMode == 0)
        rawOut[((long)(img * Co + m) * Ho + oy) * Wo + ox] = y;
      else
        nkOut[ncol * Co + m] = (_Float16)y;   // Ho=Wo=1 -> ncol == img
    }
  }
}

// ------------------------------------------------------------ dense WMMA GEMM
// out[n][m] = relu?(sum_k X[n][k]*Wt[m][k] + bias[m]) in f16. K mult of 32 (<=512),
// M mult of 16. Weight panel staged once via async-to-LDS; B = 32 contiguous
// bytes of X per lane (two b128 global loads).
__global__ void k_gemm_wmma(const _Float16* __restrict__ X, const _Float16* __restrict__ Wt,
                            const float* __restrict__ bias, _Float16* __restrict__ out,
                            int N, int M, int K, int doRelu)
{
  __shared__ alignas(16) _Float16 Wpanel[16 * 512];

  const int lane  = threadIdx.x & 31;
  const int wave  = threadIdx.x >> 5;
  const int mtile = blockIdx.y;
  const int nbase = (blockIdx.x * (int)(blockDim.x >> 5) + wave) << 4;
  const int ncol  = nbase + (lane & 15);
  const bool nvalid = ncol < N;
  const int row   = lane & 15;
  const int khalf = (lane >> 4) << 4;
  const int a_b8  = (lane >> 4) << 4;

  stage_panel(Wt + (size_t)mtile * 16 * K, Wpanel, 16 * K);
  __syncthreads();

  const _Float16* xrow = X + (size_t)(nvalid ? ncol : 0) * K;

  v8f c = {};
  for (int kk = 0; kk < K; kk += 32) {
    __builtin_prefetch(xrow + kk + 32, 0, 1);   // next B tile -> global_prefetch_b8

    v8h alo = *(const v8h*)&Wpanel[row * K + kk + a_b8];
    v8h ahi = *(const v8h*)&Wpanel[row * K + kk + 8 + a_b8];
    v16h a = __builtin_shufflevector(alo, ahi, 0,1,2,3,4,5,6,7,8,9,10,11,12,13,14,15);

    v8h blo = *(const v8h*)(xrow + kk + khalf);
    v8h bhi = *(const v8h*)(xrow + kk + khalf + 8);
    v16h b = __builtin_shufflevector(blo, bhi, 0,1,2,3,4,5,6,7,8,9,10,11,12,13,14,15);

    c = __builtin_amdgcn_wmma_f32_16x16x32_f16(false, a, false, b, (short)0, c, false, false);
  }
#pragma unroll
  for (int i = 0; i < 8; ++i) {
    int m = mtile * 16 + i + ((lane >> 4) << 3);
    if (m < M && nvalid) {
      float y = c[i] + bias[m];
      if (doRelu) y = fmaxf(y, 0.f);
      out[ncol * M + m] = (_Float16)y;
    }
  }
}

// -------------------------------------------------- BN statistics (sum,sumsq)
__global__ void k_bn_stats(const float* __restrict__ raw, float* __restrict__ stats,
                           int C, int HW, int Nimg)
{
  __shared__ float ss[256], sq[256];
  const int c = blockIdx.x;
  const int tid = threadIdx.x;
  const long total = (long)Nimg * HW;
  float s = 0.f, q = 0.f;
  for (long i = (long)blockIdx.y * 256 + tid; i < total; i += (long)gridDim.y * 256) {
    int im = (int)(i / HW);
    int sp = (int)(i - (long)im * HW);
    float v = raw[((long)im * C + c) * HW + sp];
    s += v; q += v * v;
  }
  ss[tid] = s; sq[tid] = q;
  __syncthreads();
  for (int st = 128; st > 0; st >>= 1) {
    if (tid < st) { ss[tid] += ss[tid + st]; sq[tid] += sq[tid + st]; }
    __syncthreads();
  }
  if (tid == 0) {
    atomicAdd(&stats[2 * c],     ss[0]);
    atomicAdd(&stats[2 * c + 1], sq[0]);
  }
}

// -------------------------------------- BN apply + ReLU (+ optional residual)
__global__ void k_bn_apply(const float* __restrict__ raw, const float* __restrict__ stats,
                           const float* __restrict__ g, const float* __restrict__ be,
                           const _Float16* __restrict__ skip, float skipScale,
                           _Float16* __restrict__ out, int C, int HW, int Nimg)
{
  long n = (long)Nimg * C * HW;
  long i = (long)blockIdx.x * blockDim.x + threadIdx.x;
  if (i >= n) return;
  int c = (int)((i / HW) % C);
  float cnt  = (float)Nimg * (float)HW;
  float mean = stats[2 * c] / cnt;
  float var  = stats[2 * c + 1] / cnt - mean * mean;
  float sc   = g[c] * rsqrtf(var + EPSV);
  float v = (raw[i] - mean) * sc + be[c];
  v = fmaxf(v, 0.f);
  if (skip) v += skipScale * (float)skip[i];
  out[i] = (_Float16)v;
}

// ------------------------------------------------------------- 2x2 maxpool
__global__ void k_maxpool2(const _Float16* __restrict__ in, _Float16* __restrict__ out,
                           int NC, int Hi, int Wi)
{
  int Ho = Hi >> 1, Wo = Wi >> 1;
  long n = (long)NC * Ho * Wo;
  long i = (long)blockIdx.x * blockDim.x + threadIdx.x;
  if (i >= n) return;
  int ox = (int)(i % Wo);
  int oy = (int)((i / Wo) % Ho);
  int nc = (int)(i / ((long)Wo * Ho));
  const _Float16* p = in + ((long)nc * Hi + oy * 2) * Wi + ox * 2;
  float a = (float)p[0], b = (float)p[1], c = (float)p[Wi], d = (float)p[Wi + 1];
  out[i] = (_Float16)fmaxf(fmaxf(a, b), fmaxf(c, d));
}

// ------------------------------------------- geometry MLPs (K=4 -> 32, leaky)
__global__ void k_geo_hidden(const float* __restrict__ geo,
                             const float* __restrict__ Wm1, const float* __restrict__ bm1,
                             const float* __restrict__ Wa1, const float* __restrict__ ba1,
                             float* __restrict__ hm, float* __restrict__ ha, int N)
{
  int i = blockIdx.x * blockDim.x + threadIdx.x;
  if (i >= N * 32) return;
  int n = i >> 5, j = i & 31;
  const float* x = geo + n * 4;
  float sm = bm1[j], sa = ba1[j];
#pragma unroll
  for (int k = 0; k < 4; ++k) { sm += Wm1[j * 4 + k] * x[k]; sa += Wa1[j * 4 + k] * x[k]; }
  hm[i] = sm > 0.f ? sm : 0.02f * sm;
  ha[i] = sa > 0.f ? sa : 0.02f * sa;
}

__global__ void k_geo_out(const float* __restrict__ hm, const float* __restrict__ ha,
                          const float* __restrict__ Wm2, const float* __restrict__ bm2,
                          const float* __restrict__ Wa2, const float* __restrict__ ba2,
                          float* __restrict__ gm, float* __restrict__ ga, int N)
{
  int i = blockIdx.x * blockDim.x + threadIdx.x;
  if (i >= N * 256) return;
  int n = i >> 8, j = i & 255;
  float sm = bm2[j], sa = ba2[j];
  for (int k = 0; k < 32; ++k) {
    sm += Wm2[j * 32 + k] * hm[n * 32 + k];
    sa += Wa2[j * 32 + k] * ha[n * 32 + k];
  }
  gm[i] = sm; ga[i] = sa;
}

// ----------------- embed*gm+ga, then gather [vi | partners] into mega (f16)
__global__ void k_build_mega(const _Float16* __restrict__ embed, const float* __restrict__ gm,
                             const float* __restrict__ ga, _Float16* __restrict__ mega)
{
  int i = blockIdx.x * blockDim.x + threadIdx.x;
  if (i >= 32 * 10 * 512) return;
  int b = i / 5120;
  int r = i - b * 5120;
  int d = r >> 9;
  int j = r & 511;
  int srcd = (j < 256) ? 9 : (d == 0 ? 9 : d - 1);  // order = [9,0,1,...,8]
  int jj = j & 255;
  int n = b * 10 + srcd;
  float v = (float)embed[n * 256 + jj] * gm[n * 256 + jj] + ga[n * 256 + jj];
  mega[i] = (_Float16)v;
}

// ---------------------------- final K=64 linear + (0,2,1) transpose into out
__global__ void k_final_c3(const _Float16* __restrict__ h2, const float* __restrict__ Wc3,
                           const float* __restrict__ bc3, float* __restrict__ out)
{
  int i = blockIdx.x * blockDim.x + threadIdx.x;
  if (i >= 3200) return;
  int b = i / 100;
  int r = i - b * 100;
  int o = r / 10;
  int d = r - o * 10;
  int n = b * 10 + d;
  float s = bc3[o];
  for (int k = 0; k < 64; ++k) s += Wc3[o * 64 + k] * (float)h2[n * 64 + k];
  out[i] = s;   // out[b][o][d] = big_edge[b][d][o]
}

// ============================================================================
extern "C" void kernel_launch(void* const* d_in, const int* in_sizes, int n_in,
                              void* d_out, int out_size, void* d_ws, size_t ws_size,
                              hipStream_t stream)
{
  (void)in_sizes; (void)n_in; (void)out_size; (void)ws_size;

  const float* imagelets = (const float*)d_in[0];   // (320,3,96,96)
  const float* geometry  = (const float*)d_in[1];   // (32,10,4)
  const float* W0 = (const float*)d_in[2],  *B0 = (const float*)d_in[3],
             * G0 = (const float*)d_in[4],  *BE0= (const float*)d_in[5];
  const float* W1 = (const float*)d_in[6],  *B1 = (const float*)d_in[7],
             * G1 = (const float*)d_in[8],  *BE1= (const float*)d_in[9];
  const float* W2 = (const float*)d_in[10], *B2 = (const float*)d_in[11],
             * G2 = (const float*)d_in[12], *BE2= (const float*)d_in[13];
  const float* W3 = (const float*)d_in[14], *B3 = (const float*)d_in[15],
             * G3 = (const float*)d_in[16], *BE3= (const float*)d_in[17];
  const float* W4 = (const float*)d_in[18], *B4 = (const float*)d_in[19],
             * G4 = (const float*)d_in[20], *BE4= (const float*)d_in[21];
  const float* WR = (const float*)d_in[22], *BR = (const float*)d_in[23],
             * GR = (const float*)d_in[24], *BER= (const float*)d_in[25];
  const float* W5 = (const float*)d_in[26], *B5 = (const float*)d_in[27];
  const float* Wp  = (const float*)d_in[28], *bp  = (const float*)d_in[29];
  const float* Wm1 = (const float*)d_in[30], *bm1 = (const float*)d_in[31];
  const float* Wm2 = (const float*)d_in[32], *bm2 = (const float*)d_in[33];
  const float* Wa1 = (const float*)d_in[34], *ba1 = (const float*)d_in[35];
  const float* Wa2 = (const float*)d_in[36], *ba2 = (const float*)d_in[37];
  const float* Wc1 = (const float*)d_in[38], *bc1 = (const float*)d_in[39];
  const float* Wc2 = (const float*)d_in[40], *bc2 = (const float*)d_in[41];
  const float* Wc3 = (const float*)d_in[42], *bc3 = (const float*)d_in[43];
  float* out = (float*)d_out;

  // ---------------------------- workspace carve -----------------------------
  char* base = (char*)d_ws;
  size_t off = 0;
  auto alloc = [&](size_t bytes) -> void* {
    void* p = base + off;
    off += (bytes + 255) & ~(size_t)255;
    return p;
  };
  _Float16* imgH  = (_Float16*)alloc(8847360ull * 2);
  _Float16* wc0h  = (_Float16*)alloc(16ull * 32 * 2);       // padded [Mpad][Kp]
  _Float16* wc1h  = (_Float16*)alloc(16ull * 160 * 2);
  _Float16* wc2h  = (_Float16*)alloc(32ull * 256 * 2);
  _Float16* wc3h  = (_Float16*)alloc(64ull * 512 * 2);
  _Float16* wc4h  = (_Float16*)alloc(128ull * 1024 * 2);
  _Float16* wcrh  = (_Float16*)alloc(128ull * 1152 * 2);
  _Float16* wc5h  = (_Float16*)alloc(512ull * 2048 * 2);
  _Float16* Wph   = (_Float16*)alloc(131072ull * 2);
  _Float16* Wc1h  = (_Float16*)alloc(65536ull * 2);
  _Float16* Wc2h  = (_Float16*)alloc(8192ull * 2);
  float*    bufA  = (float*)   alloc(47185920ull * 4);   // raw conv out (max conv1)
  _Float16* F1    = (_Float16*)alloc(47185920ull * 2);   // ping f16 activations
  _Float16* F2    = (_Float16*)alloc(23592960ull * 2);   // pong f16 activations
  _Float16* feats = (_Float16*)alloc(320ull * 512 * 2);
  _Float16* embB  = (_Float16*)alloc(320ull * 256 * 2);
  float*    hmB   = (float*)   alloc(320ull * 32 * 4);
  float*    haB   = (float*)   alloc(320ull * 32 * 4);
  float*    gmB   = (float*)   alloc(320ull * 256 * 4);
  float*    gaB   = (float*)   alloc(320ull * 256 * 4);
  _Float16* megaB = (_Float16*)alloc(320ull * 512 * 2);
  _Float16* h1B   = (_Float16*)alloc(320ull * 128 * 2);
  _Float16* h2B   = (_Float16*)alloc(320ull * 64 * 2);
  float*    stat  = (float*)   alloc(1024ull * 4);

  auto cvt = [&](const float* s, _Float16* d, int n) {
    k_cvt_f16<<<(n + 255) / 256, 256, 0, stream>>>(s, d, n);
  };
  auto cvtw = [&](const float* s, _Float16* d, int Co, int K) {
    int Kp = (K + 31) & ~31, Mp = (Co + 15) & ~15;
    k_cvt_wpad<<<(Mp * Kp + 255) / 256, 256, 0, stream>>>(s, d, Co, K, Kp, Mp);
  };
  auto conv = [&](const _Float16* X, const _Float16* Wf, const float* bias,
                  int Ci, int Hi, int Wi, int Co, int Kk, int st, int pd,
                  int Ho, int Wo, int mode, _Float16* nk) {
    int K = Ci * Kk * Kk;
    int Kp = (K + 31) & ~31;
    int Ntot = 320 * Ho * Wo;
    size_t shmem = (size_t)Kp * 16 * sizeof(_Float16);   // <= 64KB (conv5)
    dim3 g((Ntot + 63) / 64, ((Co + 15) & ~15) / 16);    // 4 waves/block * 16 cols
    k_conv_wmma<<<g, 128, shmem, stream>>>(X, Wf, bias, bufA, nk, mode,
                                           Ci, Hi, Wi, Co, Kk, st, pd, Ho, Wo, 320, Kp);
  };
  auto bn = [&](const float* g_, const float* be_, _Float16* dst,
                const _Float16* skip, float ss, int C, int HW) {
    (void)hipMemsetAsync(stat, 0, (size_t)C * 2 * sizeof(float), stream);
    dim3 gs(C, 64);
    k_bn_stats<<<gs, 256, 0, stream>>>(bufA, stat, C, HW, 320);
    long tot = 320L * C * HW;
    k_bn_apply<<<(int)((tot + 255) / 256), 256, 0, stream>>>(bufA, stat, g_, be_,
                                                             skip, ss, dst, C, HW, 320);
  };
  auto gemm = [&](const _Float16* X, const _Float16* Wf, const float* bias,
                  _Float16* dst, int N, int M, int K, int relu) {
    dim3 g((N + 63) / 64, (M + 15) / 16);
    k_gemm_wmma<<<g, 128, 0, stream>>>(X, Wf, bias, dst, N, M, K, relu);
  };

  // ----------------------------- weight / input converts --------------------
  cvt(imagelets, imgH, 8847360);
  cvtw(W0, wc0h,   8,   27);  cvtw(W1, wc1h,  16,  144);
  cvtw(W2, wc2h,  32,  256);  cvtw(W3, wc3h,  64,  512);
  cvtw(W4, wc4h, 128, 1024);  cvtw(WR, wcrh, 128, 1152);
  cvtw(W5, wc5h, 512, 2048);
  cvt(Wp, Wph, 131072);  cvt(Wc1, Wc1h, 65536);  cvt(Wc2, Wc2h, 8192);

  // ----------------------------- feature net --------------------------------
  conv(imgH, wc0h, B0,  3, 96, 96,   8, 3, 1, 1, 96, 96, 0, nullptr);
  bn(G0, BE0, F2, nullptr, 0.f, 8, 96 * 96);                       // act0 -> F2
  conv(F2, wc1h, B1,  8, 96, 96,  16, 3, 1, 1, 96, 96, 0, nullptr);
  bn(G1, BE1, F1, nullptr, 0.f, 16, 96 * 96);                      // act1 -> F1
  { long n = 320L * 16 * 48 * 48;
    k_maxpool2<<<(int)((n + 255) / 256), 256, 0, stream>>>(F1, F2, 320 * 16, 96, 96); } // pool1 -> F2
  conv(F2, wc2h, B2, 16, 48, 48,  32, 4, 2, 2, 25, 25, 0, nullptr);
  bn(G2, BE2, F1, nullptr, 0.f, 32, 25 * 25);                      // act2 -> F1
  conv(F1, wc3h, B3, 32, 25, 25,  64, 4, 2, 2, 13, 13, 0, nullptr);
  bn(G3, BE3, F2, nullptr, 0.f, 64, 13 * 13);                      // act3 -> F2
  { long n = 320L * 64 * 6 * 6;
    k_maxpool2<<<(int)((n + 255) / 256), 256, 0, stream>>>(F2, F1, 320 * 64, 13, 13); } // pool3 -> F1
  conv(F1, wc4h, B4, 64,  6,  6, 128, 4, 2, 2,  4,  4, 0, nullptr);
  bn(G4, BE4, F2, nullptr, 0.f, 128, 16);                          // act4 -> F2
  conv(F2, wcrh, BR, 128, 4,  4, 128, 3, 1, 1,  4,  4, 0, nullptr);
  bn(GR, BER, F1, F2, 0.1f, 128, 16);                              // actr = relu(bn)+0.1*act4 -> F1
  conv(F1, wc5h, B5, 128, 4,  4, 512, 4, 1, 0,  1,  1, 1, feats);  // feats f16 [320][512]

  // ----------------------------- head ---------------------------------------
  gemm(feats, Wph, bp, embB, 320, 256, 512, 0);                    // embed f16 [320][256]
  k_geo_hidden<<<(320 * 32 + 255) / 256, 256, 0, stream>>>(geometry, Wm1, bm1, Wa1, ba1,
                                                           hmB, haB, 320);
  k_geo_out<<<(320 * 256 + 255) / 256, 256, 0, stream>>>(hmB, haB, Wm2, bm2, Wa2, ba2,
                                                         gmB, gaB, 320);
  k_build_mega<<<(32 * 10 * 512 + 255) / 256, 256, 0, stream>>>(embB, gmB, gaB, megaB);
  gemm(megaB, Wc1h, bc1, h1B, 320, 128, 512, 1);
  gemm(h1B,   Wc2h, bc2, h2B, 320,  64, 128, 1);
  k_final_c3<<<(3200 + 255) / 256, 256, 0, stream>>>(h2B, Wc3, bc3, out);
}
